// DirectedMessage_62646392979551
// MI455X (gfx1250) — compile-verified, matrix operand order checked by reading, not run on hardware
//
#include <hip/hip_runtime.h>

typedef __attribute__((ext_vector_type(16))) _Float16 v16h;
typedef __attribute__((ext_vector_type(8)))  _Float16 v8h;
typedef __attribute__((ext_vector_type(4)))  _Float16 v4h;
typedef __attribute__((ext_vector_type(8)))  float    v8f;

static __device__ __forceinline__ v16h cat8(v8h lo, v8h hi) {
  return __builtin_shufflevector(lo, hi, 0,1,2,3,4,5,6,7,8,9,10,11,12,13,14,15);
}

// ---------------------------------------------------------------- utilities
__global__ void zero_f32_kernel(float* __restrict__ p, int n) {
  int i = blockIdx.x * blockDim.x + threadIdx.x;
  if (i < n) p[i] = 0.f;
}

// final_w is [i][j][l] row-major == exactly the B matrix [i][k] we need.
__global__ void cvt_f32_to_f16_kernel(const float* __restrict__ src,
                                      _Float16* __restrict__ dst, int n) {
  int i = blockIdx.x * blockDim.x + threadIdx.x;
  if (i < n) dst[i] = (_Float16)src[i];
}

// -------------------------------------------------- kernel 1: per-edge me
// me[e,:] = silu(m_ji[e] @ W_nbr + b_nbr) * (e_rbf[e] @ W_e), stored f16.
// Block: 256 threads = 8 waves; tile 128 edges x 128 cols, K = 128.
__global__ __launch_bounds__(256) void edge_me_kernel(
    const float* __restrict__ m_ji, const float* __restrict__ e_rbf,
    const float* __restrict__ W_nbr, const float* __restrict__ b_nbr,
    const float* __restrict__ W_e, _Float16* __restrict__ me_h, int n_edges)
{
  __shared__ __align__(16) _Float16 As[128 * 136];  // [edge][k]
  __shared__ __align__(16) _Float16 Bt[128 * 136];  // [n][k] (W_nbr transposed)
  __shared__ float Er[128 * 6];

  const int tid = threadIdx.x;
  const int e0  = blockIdx.x * 128;

  for (int idx = tid; idx < 4096; idx += 256) {     // m_ji tile, b128 loads
    int e = idx >> 5, l0 = (idx & 31) * 4;
    int ge = e0 + e;
    float4 v = make_float4(0.f, 0.f, 0.f, 0.f);
    if (ge < n_edges) v = *(const float4*)&m_ji[(size_t)ge * 128 + l0];
    v4h h = {(_Float16)v.x, (_Float16)v.y, (_Float16)v.z, (_Float16)v.w};
    *(v4h*)&As[e * 136 + l0] = h;
  }
  for (int idx = tid; idx < 4096; idx += 256) {     // W_nbr transpose, b128 loads
    int k = idx >> 5, n0 = (idx & 31) * 4;
    float4 v = *(const float4*)&W_nbr[k * 128 + n0];
    Bt[(n0 + 0) * 136 + k] = (_Float16)v.x;
    Bt[(n0 + 1) * 136 + k] = (_Float16)v.y;
    Bt[(n0 + 2) * 136 + k] = (_Float16)v.z;
    Bt[(n0 + 3) * 136 + k] = (_Float16)v.w;
  }
  for (int idx = tid; idx < 128 * 6; idx += 256) {
    int ge = e0 + idx / 6;
    Er[idx] = (ge < n_edges) ? e_rbf[(size_t)ge * 6 + (idx % 6)] : 0.f;
  }
  __syncthreads();

  const int lane = tid & 31;
  const int m0   = (tid >> 5) * 16;        // wave's 16 rows
  const int lrow = lane & 15;
  const int hsel = (lane >> 4) << 3;       // K-half select per WMMA f16 layout

  v8f acc[8];
  const v8f vz = {0.f,0.f,0.f,0.f,0.f,0.f,0.f,0.f};
  #pragma unroll
  for (int t = 0; t < 8; ++t) acc[t] = vz;

  #pragma unroll
  for (int kk = 0; kk < 128; kk += 32) {
    const _Float16* ap = &As[(m0 + lrow) * 136 + kk + hsel];
    v16h a = cat8(*(const v8h*)ap, *(const v8h*)(ap + 16));
    #pragma unroll
    for (int t = 0; t < 8; ++t) {
      const _Float16* bp = &Bt[(t * 16 + lrow) * 136 + kk + hsel];
      v16h b = cat8(*(const v8h*)bp, *(const v8h*)(bp + 16));
      acc[t] = __builtin_amdgcn_wmma_f32_16x16x32_f16(false, a, false, b,
                                                      (short)0, acc[t], false, false);
    }
  }

  const int rup = (lane >> 4) << 3;        // C/D layout: lanes 16-31 hold M+8
  #pragma unroll
  for (int t = 0; t < 8; ++t) {
    int col = t * 16 + (lane & 15);
    float bias = b_nbr[col];
    float w0 = W_e[col],       w1 = W_e[128 + col], w2 = W_e[256 + col],
          w3 = W_e[384 + col], w4 = W_e[512 + col], w5 = W_e[640 + col];
    #pragma unroll
    for (int r = 0; r < 8; ++r) {
      int rl = m0 + rup + r;
      float x = acc[t][r] + bias;
      float sil = x / (1.f + __expf(-x));
      const float* er = &Er[rl * 6];
      float g = er[0]*w0 + er[1]*w1 + er[2]*w2 + er[3]*w3 + er[4]*w4 + er[5]*w5;
      int ge = e0 + rl;
      if (ge < n_edges) me_h[(size_t)ge * 128 + col] = (_Float16)(sil * g);
    }
  }
}

// ------------------------------------ kernel 2: A_seg = segsum(a_sbf @ W_a)
// LDS-staged so the 196 MB a_sbf stream is fully coalesced; 8 f32 atomics/angle
// into a 6.4 MB L2-resident table.
__global__ __launch_bounds__(128) void angle_scatter_kernel(
    const float* __restrict__ a_sbf, const int* __restrict__ kj_idx,
    const float* __restrict__ W_a, float* __restrict__ A_seg, int n_angles)
{
  __shared__ float sa[128 * 49];
  __shared__ float wa[49 * 8];
  const int tid = threadIdx.x;
  const long base = (long)blockIdx.x * 128;
  const long lim  = (long)n_angles * 49;

  for (int idx = tid; idx < 49 * 8; idx += 128) wa[idx] = W_a[idx];
  for (int idx = tid; idx < 128 * 49; idx += 128) {
    long g = base * 49 + idx;
    sa[idx] = (g < lim) ? a_sbf[g] : 0.f;
  }
  __syncthreads();

  long w = base + tid;
  if (w < n_angles) {
    float s[8];
    #pragma unroll
    for (int j = 0; j < 8; ++j) s[j] = 0.f;
    const float* row = &sa[tid * 49];
    for (int q = 0; q < 49; ++q) {
      float av = row[q];
      #pragma unroll
      for (int j = 0; j < 8; ++j) s[j] += av * wa[q * 8 + j];
    }
    int kj = kj_idx[w];
    float* dst = &A_seg[(size_t)kj * 8];
    #pragma unroll
    for (int j = 0; j < 8; ++j) atomicAdd(&dst[j], s[j]);
  }
}

// ----------------- kernel 3: final[e,i] = sum_{j,l} Aseg[e,j]*me[e,l]*T[i,j,l]
// GEMM [E,1024]@[1024,128]; LHS rows formed on the fly (me frag scaled by Aseg).
// B chunks double-buffered via GLOBAL_LOAD_ASYNC_TO_LDS_B128 (ASYNCcnt path).

// Issue one 128x128-halves chunk (32 KB) of T2h into LDS asynchronously.
// 2048 16-byte transfers / 256 threads = 8 async ops per thread.
static __device__ __forceinline__ void issue_bt_chunk(
    const _Float16* __restrict__ T2h, _Float16* bt_base, int j, int tid)
{
  #pragma unroll
  for (int s = 0; s < 8; ++s) {
    int c  = tid + 256 * s;          // 16B-chunk id, 0..2047
    int i  = c >> 4;                 // output column i (row of Bt)
    int l0 = (c & 15) * 8;           // 8 halves = 16 bytes
    unsigned lds = (unsigned)(unsigned long long)&bt_base[i * 136 + l0];
    const _Float16* g = &T2h[(size_t)i * 1024 + j * 128 + l0];
    asm volatile("global_load_async_to_lds_b128 %0, %1, off"
                 :: "v"(lds), "v"(g) : "memory");
  }
}

__global__ __launch_bounds__(256) void final_gemm_kernel(
    const _Float16* __restrict__ me_h, const float* __restrict__ A_seg,
    const _Float16* __restrict__ T2h, float* __restrict__ out, int n_edges)
{
  __shared__ __align__(16) _Float16 Ms[128 * 136];      // me tile (all j reuse)
  __shared__ __align__(16) _Float16 Bt[2][128 * 136];   // double-buffered chunk
  __shared__ float Asg[128 * 8];

  const int tid = threadIdx.x;
  const int e0  = blockIdx.x * 128;

  // kick off chunk 0 while we stage Ms/Asg
  issue_bt_chunk(T2h, Bt[0], 0, tid);

  for (int idx = tid; idx < 2048; idx += 256) {   // me tile, b128 loads
    int e = idx >> 4, l0 = (idx & 15) * 8;
    int ge = e0 + e;
    v8h pk = {(_Float16)0, (_Float16)0, (_Float16)0, (_Float16)0,
              (_Float16)0, (_Float16)0, (_Float16)0, (_Float16)0};
    if (ge < n_edges) pk = *(const v8h*)&me_h[(size_t)ge * 128 + l0];
    *(v8h*)&Ms[e * 136 + l0] = pk;
  }
  for (int idx = tid; idx < 1024; idx += 256) {
    int ge = e0 + (idx >> 3);
    Asg[idx] = (ge < n_edges) ? A_seg[(size_t)ge * 8 + (idx & 7)] : 0.f;
  }

  asm volatile("s_wait_asynccnt 0x0" ::: "memory");
  __syncthreads();

  const int lane = tid & 31;
  const int m0   = (tid >> 5) * 16;
  const int lrow = lane & 15;
  const int hsel = (lane >> 4) << 3;

  v8f acc[8];
  const v8f vz = {0.f,0.f,0.f,0.f,0.f,0.f,0.f,0.f};
  #pragma unroll
  for (int t = 0; t < 8; ++t) acc[t] = vz;

  int p = 0;
  for (int j = 0; j < 8; ++j) {
    // prefetch next chunk into the buffer everyone finished reading last iter
    if (j + 1 < 8) issue_bt_chunk(T2h, Bt[p ^ 1], j + 1, tid);

    const _Float16* bt = Bt[p];
    _Float16 aj = (_Float16)Asg[(m0 + lrow) * 8 + j];
    v8h ajv = {aj, aj, aj, aj, aj, aj, aj, aj};
    #pragma unroll
    for (int kk = 0; kk < 128; kk += 32) {
      const _Float16* ap = &Ms[(m0 + lrow) * 136 + kk + hsel];
      v8h alo = (*(const v8h*)ap) * ajv;
      v8h ahi = (*(const v8h*)(ap + 16)) * ajv;
      v16h a = cat8(alo, ahi);
      #pragma unroll
      for (int t = 0; t < 8; ++t) {
        const _Float16* bp = &bt[(t * 16 + lrow) * 136 + kk + hsel];
        v16h b = cat8(*(const v8h*)bp, *(const v8h*)(bp + 16));
        acc[t] = __builtin_amdgcn_wmma_f32_16x16x32_f16(false, a, false, b,
                                                        (short)0, acc[t], false, false);
      }
    }

    asm volatile("s_wait_asynccnt 0x0" ::: "memory");
    __syncthreads();
    p ^= 1;
  }

  const int rup = (lane >> 4) << 3;
  #pragma unroll
  for (int t = 0; t < 8; ++t) {
    int col = t * 16 + (lane & 15);
    #pragma unroll
    for (int r = 0; r < 8; ++r) {
      int ge = e0 + m0 + rup + r;
      if (ge < n_edges) out[(size_t)ge * 128 + col] = acc[t][r];
    }
  }
}

// ---------------------------------------------------------------- launcher
extern "C" void kernel_launch(void* const* d_in, const int* in_sizes, int n_in,
                              void* d_out, int out_size, void* d_ws, size_t ws_size,
                              hipStream_t stream) {
  const float* m_ji    = (const float*)d_in[0];
  const float* e_rbf   = (const float*)d_in[3];
  const float* a_sbf   = (const float*)d_in[4];
  const int*   kj_idx  = (const int*)d_in[5];
  const float* W_nbr   = (const float*)d_in[6];
  const float* b_nbr   = (const float*)d_in[7];
  const float* W_e     = (const float*)d_in[8];
  const float* W_a     = (const float*)d_in[9];
  const float* final_w = (const float*)d_in[10];

  const int n_edges  = in_sizes[0] / 128;
  const int n_angles = in_sizes[5];

  // workspace layout
  char* ws = (char*)d_ws;
  _Float16* me_h = (_Float16*)ws;
  size_t off = ((size_t)n_edges * 128 * sizeof(_Float16) + 255) & ~(size_t)255;
  float* A_seg = (float*)(ws + off);
  off += ((size_t)n_edges * 8 * sizeof(float) + 255) & ~(size_t)255;
  _Float16* T2h = (_Float16*)(ws + off);   // 128*8*128 halves

  const int nT = 128 * 8 * 128;
  zero_f32_kernel<<<(n_edges * 8 + 255) / 256, 256, 0, stream>>>(A_seg, n_edges * 8);
  cvt_f32_to_f16_kernel<<<(nT + 255) / 256, 256, 0, stream>>>(final_w, T2h, nT);

  const int blocksE = (n_edges + 127) / 128;
  edge_me_kernel<<<blocksE, 256, 0, stream>>>(m_ji, e_rbf, W_nbr, b_nbr, W_e,
                                              me_h, n_edges);
  angle_scatter_kernel<<<(n_angles + 127) / 128, 128, 0, stream>>>(
      a_sbf, kj_idx, W_a, A_seg, n_angles);
  final_gemm_kernel<<<blocksE, 256, 0, stream>>>(me_h, A_seg, T2h,
                                                 (float*)d_out, n_edges);
}
